// GATSparse_627065225300
// MI455X (gfx1250) — matrix operand order repaired and checked by reading, not compiled, not run on hardware
//
#include <hip/hip_runtime.h>
#include <hip/hip_bf16.h>

// Problem constants (match reference setup_inputs).
#define GN   32768   // nodes per batch
#define GD   128     // feature dim
#define GE   262144  // edges per batch
#define GH   8       // heads
#define GZD  256     // concat dim
#define GOUT 128     // output channels
#define GB   2       // batch
#define NTILES 17    // 8 (Wm) + 8 (Wskip) + 1 ([Wa1|Wa2])

typedef __attribute__((ext_vector_type(16))) __bf16 v16bf;
typedef __attribute__((ext_vector_type(8)))  float  v8f;

__device__ __forceinline__ float lrelu(float x) { return x > 0.f ? x : 0.01f * x; }

// IEEE float atomic max via integer atomics (monotone transform).
__device__ __forceinline__ void atomicMaxF(float* addr, float val) {
  if (val >= 0.f) atomicMax((int*)addr, __float_as_int(val));
  else            atomicMin((unsigned int*)addr, __float_as_uint(val));
}

// ---------------------------------------------------------------------------
// One-time weight pre-swizzle: fp32 W -> bf16 fragments in WMMA B-operand
// register order: wf[((tile*8 + kc)*32 + lane)*16 + 2i (+1)]
//   B fragment 32x16 (16-bit): lane ln holds col N=ln, half hs=lane/16,
//   VGPR i holds K = kc*32 + (i/4)*16 + hs*8 + 2(i%4) (lo) and +1 (hi).
// tiles 0..7  : Wm  cols [16t,16t+16)
// tiles 8..15 : Wsk cols [16(t-8),...)
// tile  16    : cols 0..7 from Wa1, 8..15 from Wa2
// ---------------------------------------------------------------------------
__global__ void wfrag_kernel(const float* __restrict__ Wm,  const float* __restrict__ Wsk,
                             const float* __restrict__ Wa1, const float* __restrict__ Wa2,
                             __bf16* __restrict__ wf)
{
  const int t = blockIdx.x * blockDim.x + threadIdx.x;   // NTILES*8*32 = 4352
  if (t >= NTILES * 8 * 32) return;
  const int lane = t & 31;
  const int kc   = (t >> 5) & 7;
  const int tile = t >> 8;
  const int hs = lane >> 4, ln = lane & 15;
  __bf16* dst = wf + (size_t)t * 16;
#pragma unroll
  for (int i = 0; i < 8; ++i) {
    const int k = kc * 32 + (i >> 2) * 16 + hs * 8 + 2 * (i & 3);
    float w0, w1;
    if (tile < 8) {
      const int col = tile * 16 + ln;
      w0 = Wm[(size_t)k * GOUT + col];  w1 = Wm[(size_t)(k + 1) * GOUT + col];
    } else if (tile < 16) {
      const int col = (tile - 8) * 16 + ln;
      w0 = Wsk[(size_t)k * GOUT + col]; w1 = Wsk[(size_t)(k + 1) * GOUT + col];
    } else if (ln < GH) {
      w0 = Wa1[(size_t)k * GH + ln];    w1 = Wa1[(size_t)(k + 1) * GH + ln];
    } else {
      w0 = Wa2[(size_t)k * GH + (ln - GH)]; w1 = Wa2[(size_t)(k + 1) * GH + (ln - GH)];
    }
    dst[2 * i]     = (__bf16)w0;
    dst[2 * i + 1] = (__bf16)w1;
  }
}

// ---------------------------------------------------------------------------
// z = concat(nf, hid) converted to bf16, row-major [N, 256].
// One thread per packed pair (both sources lie in the same half: cols even).
// ---------------------------------------------------------------------------
__global__ void zconv_kernel(const float* __restrict__ nf, const float* __restrict__ hid,
                             __bf16* __restrict__ zb)
{
  const int t = blockIdx.x * blockDim.x + threadIdx.x;   // GN * GZD/2
  if (t >= GN * (GZD / 2)) return;
  const int n = t >> 7;            // GZD/2 = 128
  const int c = (t & 127) * 2;
  float2 f;
  if (c < GD) f = *(const float2*)(nf  + (size_t)n * GD + c);
  else        f = *(const float2*)(hid + (size_t)n * GD + (c - GD));
  __bf16* zr = zb + (size_t)n * GZD + c;
  zr[0] = (__bf16)f.x;
  zr[1] = (__bf16)f.y;
}

// ---------------------------------------------------------------------------
// Dense node projections with WMMA (bf16 -> f32 accum). One wave per 16 rows.
// A fragments come straight off bf16 z rows (two contiguous 16B runs per
// K-chunk); B fragments are single 32B pre-swizzled loads. No converts.
// Outputs: v = z@Wm+bm, agg = z@Wskip+bskip (residual pre-seed),
//          att1/att2 = z@[Wa1|Wa2]+bias.
// ---------------------------------------------------------------------------
__global__ void __launch_bounds__(128)
dense_wmma_kernel(const __bf16* __restrict__ zb, const __bf16* __restrict__ wf,
                  const float* __restrict__ bm,  const float* __restrict__ bsk,
                  const float* __restrict__ ba1, const float* __restrict__ ba2,
                  float* __restrict__ v, float* __restrict__ agg,
                  float* __restrict__ att1, float* __restrict__ att2)
{
  const int lane = threadIdx.x & 31;
  const int wave = threadIdx.x >> 5;
  const int m0   = (blockIdx.x * 4 + wave) * 16;   // 4 waves/block, 16 rows/wave
  const int hs   = lane >> 4;
  const int ln   = lane & 15;
  const int row  = m0 + ln;                        // A: lane holds row M = lane%16

  // ---- A fragments: VGPRs 0..3 hold K=kc*32+hs*8..+7, VGPRs 4..7 hold +16 ----
  const __bf16* zr = zb + (size_t)row * GZD;
  v16bf a[8];
#pragma unroll
  for (int kc = 0; kc < 8; ++kc) {
    union { uint4 u[2]; v16bf v; } ab;
    ab.u[0] = *(const uint4*)(zr + kc * 32 + hs * 8);
    ab.u[1] = *(const uint4*)(zr + kc * 32 + 16 + hs * 8);
    a[kc] = ab.v;
  }

  const v16bf* wfv = (const v16bf*)wf;

  // ---- 17 output column tiles, K=256 in 8 WMMA steps each ----
  for (int t = 0; t < NTILES; ++t) {
    v8f acc = {};
#pragma unroll
    for (int kc = 0; kc < 8; ++kc) {
      const v16bf bfr = wfv[(size_t)((t * 8 + kc) << 5) + lane];
      acc = __builtin_amdgcn_wmma_f32_16x16x32_bf16(
                false, a[kc], false, bfr, (short)0, acc, false, false);
    }
    if (t < 16) {
      float* out        = (t < 8) ? v : agg;
      const float* bias = (t < 8) ? bm : bsk;
      const int col     = (t & 7) * 16 + ln;
      const float bv = bias[col];
#pragma unroll
      for (int r = 0; r < 8; ++r)                  // C: row = m0 + hs*8 + r
        out[(size_t)(m0 + hs * 8 + r) * GOUT + col] = acc[r] + bv;
    } else {
      const float bv = (ln < GH) ? ba1[ln] : ba2[ln - GH];
#pragma unroll
      for (int r = 0; r < 8; ++r) {
        const int rw = m0 + hs * 8 + r;
        if (ln < GH) att1[(size_t)rw * GH + ln]        = acc[r] + bv;
        else         att2[(size_t)rw * GH + (ln - GH)] = acc[r] + bv;
      }
    }
  }
}

// ---------------------------------------------------------------------------
// Edge-feature logits (gkt stage only): eatt[e,h] = edge_fts[e,:]@Wae[:,h]+bae
// ---------------------------------------------------------------------------
__global__ void edge_att_kernel(const float* __restrict__ efts,
                                const float* __restrict__ Wae,
                                const float* __restrict__ bae,
                                float* __restrict__ eatt)
{
  const int t = blockIdx.x * blockDim.x + threadIdx.x;
  if (t >= GE * GH) return;
  const int e = t >> 3, h = t & 7;
  const float* rowp = efts + (size_t)e * GD;
  float s = bae[h];
#pragma unroll 8
  for (int k = 0; k < GD; ++k) s += rowp[k] * Wae[k * GH + h];
  eatt[t] = s;
}

// nodemax = -inf, nodesum = 0
__global__ void init_node_kernel(float* __restrict__ nmax, float* __restrict__ nsum)
{
  const int t = blockIdx.x * blockDim.x + threadIdx.x;
  if (t < GN * GH) { nmax[t] = -3.402823466e38f; nsum[t] = 0.f; }
}

// logits = leaky_relu(att1[src] + att2[tgt] (+ eatt)); segment max over tgt
__global__ void edge_logit_max_kernel(const int* __restrict__ idx,
                                      const float* __restrict__ att1,
                                      const float* __restrict__ att2,
                                      const float* __restrict__ eatt,
                                      float* __restrict__ elog,
                                      float* __restrict__ nmax)
{
  const int t = blockIdx.x * blockDim.x + threadIdx.x;
  if (t >= GE * GH) return;
  const int e = t >> 3, h = t & 7;
  const int src = idx[2 * e], tgt = idx[2 * e + 1];
  float l = att1[src * GH + h] + att2[tgt * GH + h];
  if (eatt) l += eatt[t];
  l = lrelu(l);
  elog[t] = l;
  atomicMaxF(nmax + tgt * GH + h, l);
}

// e = exp(logit - max[tgt]); segment sum over tgt (overwrites elog with exp)
__global__ void edge_exp_sum_kernel(const int* __restrict__ idx,
                                    float* __restrict__ elog,
                                    const float* __restrict__ nmax,
                                    float* __restrict__ nsum)
{
  const int t = blockIdx.x * blockDim.x + threadIdx.x;
  if (t >= GE * GH) return;
  const int e = t >> 3, h = t & 7;
  const int tgt = idx[2 * e + 1];
  const float ex = __expf(elog[t] - nmax[tgt * GH + h]);
  elog[t] = ex;
  atomicAdd(nsum + tgt * GH + h, ex);
}

// agg[tgt, c] += (exp/sum) * v[src, c]  (agg pre-seeded with skip branch)
__global__ void scatter_kernel(const int* __restrict__ idx,
                               const float* __restrict__ eexp,
                               const float* __restrict__ nsum,
                               const float* __restrict__ v,
                               float* __restrict__ agg)
{
  const int t = blockIdx.x * blockDim.x + threadIdx.x;
  if (t >= GE * GOUT) return;
  const int e = t >> 7, c = t & 127, h = c >> 4;
  const int src = idx[2 * e], tgt = idx[2 * e + 1];
  const float coef = eexp[e * GH + h] / nsum[tgt * GH + h];
  atomicAdd(agg + (size_t)tgt * GOUT + c, coef * v[(size_t)src * GOUT + c]);
}

// dst = relu(agg)
__global__ void relu_out_kernel(const float* __restrict__ agg, float* __restrict__ dst)
{
  const int t = blockIdx.x * blockDim.x + threadIdx.x;
  if (t < GN * GOUT) dst[t] = fmaxf(agg[t], 0.f);
}

// ---------------------------------------------------------------------------
extern "C" void kernel_launch(void* const* d_in, const int* in_sizes, int n_in,
                              void* d_out, int out_size, void* d_ws, size_t ws_size,
                              hipStream_t stream)
{
  const float* node_fts = (const float*)d_in[0];   // [B,N,D]
  const float* gkt_efts = (const float*)d_in[1];   // [B,E,D]
  const float* hidden   = (const float*)d_in[2];   // [B,N,D]
  const int*   cfg_idx  = (const int*)d_in[3];     // [B,E,2]
  const int*   gkt_idx  = (const int*)d_in[4];     // [B,E,2]
  const float* Wm  = (const float*)d_in[5];
  const float* bm  = (const float*)d_in[6];
  const float* Wsk = (const float*)d_in[7];
  const float* bsk = (const float*)d_in[8];
  const float* Wa1 = (const float*)d_in[9];
  const float* ba1 = (const float*)d_in[10];
  const float* Wa2 = (const float*)d_in[11];
  const float* ba2 = (const float*)d_in[12];
  const float* Wae = (const float*)d_in[13];
  const float* bae = (const float*)d_in[14];
  float* out = (float*)d_out;                      // [B,N,OUT]

  // Workspace layout (float-sized units). Total ~22M floats (~88 MB).
  float* ws = (float*)d_ws;
  size_t o = 0;
  float*  v    = ws + o; o += (size_t)GN * GOUT;   // per-head values
  float*  agg  = ws + o; o += (size_t)GN * GOUT;   // skip + aggregated messages
  float*  att1 = ws + o; o += (size_t)GN * GH;
  float*  att2 = ws + o; o += (size_t)GN * GH;
  float*  nmax = ws + o; o += (size_t)GN * GH;
  float*  nsum = ws + o; o += (size_t)GN * GH;
  float*  elog = ws + o; o += (size_t)GE * GH;     // logits, then exp
  float*  eatt = ws + o; o += (size_t)GE * GH;     // gkt edge logits
  float*  cfgh = ws + o; o += (size_t)GN * GOUT;   // cfg-stage hidden output
  __bf16* zb   = (__bf16*)(ws + o); o += (size_t)GN * GZD / 2;        // bf16 z
  __bf16* wf   = (__bf16*)(ws + o); o += (size_t)NTILES * 8 * 32 * 8; // bf16 W frags

  const int denseBlocks = GN / 64;                 // 4 waves * 16 rows per block
  const int nhBlocks    = (GN * GH + 255) / 256;
  const int ehBlocks    = (GE * GH + 255) / 256;
  const int ecBlocks    = (GE * GOUT) / 256;
  const int ncBlocks    = (GN * GOUT) / 256;
  const int zcBlocks    = (GN * (GZD / 2)) / 256;
  const int wfBlocks    = (NTILES * 8 * 32 + 255) / 256;

  // One-time weight pre-swizzle (bf16, fragment order).
  wfrag_kernel<<<wfBlocks, 256, 0, stream>>>(Wm, Wsk, Wa1, Wa2, wf);

  auto run_stage = [&](const float* nf_b, const float* hid_b, const int* idx_b,
                       const float* eatt_or_null, float* dst_b) {
    zconv_kernel<<<zcBlocks, 256, 0, stream>>>(nf_b, hid_b, zb);
    dense_wmma_kernel<<<denseBlocks, 128, 0, stream>>>(
        zb, wf, bm, bsk, ba1, ba2, v, agg, att1, att2);
    init_node_kernel<<<nhBlocks, 256, 0, stream>>>(nmax, nsum);
    edge_logit_max_kernel<<<ehBlocks, 256, 0, stream>>>(idx_b, att1, att2,
                                                        eatt_or_null, elog, nmax);
    edge_exp_sum_kernel<<<ehBlocks, 256, 0, stream>>>(idx_b, elog, nmax, nsum);
    scatter_kernel<<<ecBlocks, 256, 0, stream>>>(idx_b, elog, nsum, v, agg);
    relu_out_kernel<<<ncBlocks, 256, 0, stream>>>(agg, dst_b);
  };

  for (int b = 0; b < GB; ++b) {
    const float* nf_b  = node_fts + (size_t)b * GN * GD;
    const float* hid_b = hidden   + (size_t)b * GN * GD;
    // ---- cfg stage: z=[nf|hidden], cfg edges, no extra logits -> cfgh ----
    run_stage(nf_b, hid_b, cfg_idx + (size_t)b * GE * 2, nullptr, cfgh);
    // ---- gkt stage: edge logits from gkt_edge_fts, z=[nf|cfgh] -> out ----
    edge_att_kernel<<<ehBlocks, 256, 0, stream>>>(
        gkt_efts + (size_t)b * GE * GD, Wae, bae, eatt);
    run_stage(nf_b, cfgh, gkt_idx + (size_t)b * GE * 2, eatt,
              out + (size_t)b * GN * GOUT);
  }
}
